// SearchTransfer_43061342110120
// MI455X (gfx1250) — matrix operand order repaired and checked by reading, not compiled, not run on hardware
//
#include <hip/hip_runtime.h>
#include <math.h>

typedef __attribute__((ext_vector_type(16))) _Float16 v16h;
typedef __attribute__((ext_vector_type(8)))  float    v8f;
typedef __attribute__((ext_vector_type(4)))  unsigned u4v;

struct U8x { u4v lo, hi; };   // 32 bytes == v16h

#define BSZ   2
#define LDIM  2304          // 48*48 columns; also F = 256*9
#define HH    48
#define APAD  72            // padded LDS row stride (halves): 36 dwords, bank-conflict free

// ---------------------------------------------------------------------------
// Kernel 1: unfold(3,1,1) + column L2-normalize -> f16, row-major [b][col][f]
// One block per (b, col). 256 threads, 9 f-elements each (9*256 = 2304).
// ---------------------------------------------------------------------------
__global__ void __launch_bounds__(256)
unfold_norm_kernel(const float* __restrict__ lr, const float* __restrict__ refsr,
                   _Float16* __restrict__ QnT, _Float16* __restrict__ KnT)
{
    const int blk = blockIdx.x;           // b*LDIM + l
    const int b = blk / LDIM;
    const int l = blk % LDIM;
    const int y = l / HH, x = l % HH;
    const int tid = threadIdx.x;

    float qv[9], kv[9];
    float sq = 0.f, sk = 0.f;
#pragma unroll
    for (int j = 0; j < 9; ++j) {
        const int f = tid + j * 256;
        const int c = f / 9, r = f % 9;
        const int di = r / 3, dj = r % 3;
        const int sy = y + di - 1, sx = x + dj - 1;
        float q = 0.f, k = 0.f;
        if ((unsigned)sy < 48u && (unsigned)sx < 48u) {
            const size_t off = (((size_t)b * 256 + c) * 48 + sy) * 48 + sx;
            q = lr[off];
            k = refsr[off];
        }
        qv[j] = q; kv[j] = k;
        sq += q * q; sk += k * k;
    }
    __shared__ float redq[256], redk[256];
    redq[tid] = sq; redk[tid] = sk;
    __syncthreads();
    for (int s = 128; s > 0; s >>= 1) {
        if (tid < s) { redq[tid] += redq[tid + s]; redk[tid] += redk[tid + s]; }
        __syncthreads();
    }
    const float invq = 1.f / fmaxf(sqrtf(redq[0]), 1e-12f);
    const float invk = 1.f / fmaxf(sqrtf(redk[0]), 1e-12f);
    const size_t base = ((size_t)b * LDIM + l) * LDIM;
#pragma unroll
    for (int j = 0; j < 9; ++j) {
        const int f = tid + j * 256;
        QnT[base + f] = (_Float16)(qv[j] * invq);
        KnT[base + f] = (_Float16)(kv[j] * invk);
    }
}

// ---------------------------------------------------------------------------
// Kernel 2: R[b][l][m] = sum_f KnT[b][l][f] * QnT[b][m][f]
// Block tile: 256 (l) x 64 (m), K-stage 64.  8 waves; wave w owns a 32x64 tile
// (2 A-fragments x 4 B-fragments -> 8 WMMAs per 32-K substep, B reused).
// LDS rows padded to 72 halves to kill bank conflicts on fragment loads.
// ---------------------------------------------------------------------------
__global__ void __launch_bounds__(256)
corr_gemm_kernel(const _Float16* __restrict__ KnT, const _Float16* __restrict__ QnT,
                 float* __restrict__ R)
{
    const int m0   = blockIdx.x * 64;
    const int l0   = blockIdx.y * 256;
    const int b    = blockIdx.z;
    const int tid  = threadIdx.x;
    const int wave = tid >> 5;
    const int lane = tid & 31;
    const int hlane = lane & 15;
    const bool low = lane < 16;

    __shared__ __align__(16) _Float16 ldsA[256 * APAD];   // 36 KB
    __shared__ __align__(16) _Float16 ldsB[64 * APAD];    //  9 KB

    v8f z = {};
    v8f acc[2][4] = { { z, z, z, z }, { z, z, z, z } };

    const _Float16* Abase = KnT + ((size_t)b * LDIM + l0) * LDIM;
    const _Float16* Bbase = QnT + ((size_t)b * LDIM + m0) * LDIM;

    for (int k0 = 0; k0 < LDIM; k0 += 64) {
        __syncthreads();
        // A tile: 256 rows x 64 halves  (2048 x 16B chunks, 8 per thread)
#pragma unroll
        for (int it = 0; it < 8; ++it) {
            const int u = tid + it * 256;
            const int r = u >> 3, part = u & 7;
            *((u4v*)&ldsA[r * APAD + part * 8]) =
                *((const u4v*)(Abase + (size_t)r * LDIM + k0 + part * 8));
        }
        // B tile: 64 rows x 64 halves  (512 x 16B chunks, 2 per thread)
#pragma unroll
        for (int it = 0; it < 2; ++it) {
            const int u = tid + it * 256;
            const int r = u >> 3, part = u & 7;
            *((u4v*)&ldsB[r * APAD + part * 8]) =
                *((const u4v*)(Bbase + (size_t)r * LDIM + k0 + part * 8));
        }
        __syncthreads();

#pragma unroll
        for (int kk = 0; kk < 64; kk += 32) {
            // A fragments (16x32 f16): lane<16 -> K 0-7 & 16-23, lane>=16 -> K 8-15 & 24-31
            v16h afrag[2];
#pragma unroll
            for (int as = 0; as < 2; ++as) {
                U8x au;
                const int row = wave * 32 + as * 16 + hlane;
                const int kb = kk + (low ? 0 : 8);
                au.lo = *((const u4v*)&ldsA[row * APAD + kb]);
                au.hi = *((const u4v*)&ldsA[row * APAD + kb + 16]);
                afrag[as] = __builtin_bit_cast(v16h, au);
            }
#pragma unroll
            for (int ms = 0; ms < 4; ++ms) {
                // B fragment (32x16 f16): lane<16 -> col N=lane, K 0-7 & 16-23; lane>=16 -> K 8-15 & 24-31
                U8x bu;
                const int n = ms * 16 + hlane;
                const int kb = kk + (low ? 0 : 8);
                bu.lo = *((const u4v*)&ldsB[n * APAD + kb]);
                bu.hi = *((const u4v*)&ldsB[n * APAD + kb + 16]);
                const v16h bfrag = __builtin_bit_cast(v16h, bu);
#pragma unroll
                for (int as = 0; as < 2; ++as)
                    acc[as][ms] = __builtin_amdgcn_wmma_f32_16x16x32_f16(
                        false, afrag[as], false, bfrag, (short)0, acc[as][ms], false, false);
            }
        }
    }
    // C/D layout: VGPR r, lane<16 -> M=r, lane>=16 -> M=r+8; N = lane&15
#pragma unroll
    for (int as = 0; as < 2; ++as) {
        const int lrow = l0 + wave * 32 + as * 16 + (low ? 0 : 8);
#pragma unroll
        for (int ms = 0; ms < 4; ++ms) {
            const int gm = m0 + ms * 16 + hlane;
#pragma unroll
            for (int r = 0; r < 8; ++r)
                R[((size_t)b * LDIM + (lrow + r)) * LDIM + gm] = acc[as][ms][r];
        }
    }
}

// ---------------------------------------------------------------------------
// Kernel 3: H_idx[b][m] = argmax_l R[b][l][m]   (first max wins, like jnp.argmax)
// ---------------------------------------------------------------------------
__global__ void argmax_kernel(const float* __restrict__ R, int* __restrict__ Hidx)
{
    const int t = blockIdx.x * 256 + threadIdx.x;
    if (t >= BSZ * LDIM) return;
    const int b = t / LDIM, m = t % LDIM;
    const float* col = R + (size_t)b * LDIM * LDIM + m;
    float best = col[0];
    int bi = 0;
    for (int l = 1; l < LDIM; ++l) {
        const float v = col[(size_t)l * LDIM];
        if (v > best) { best = v; bi = l; }
    }
    Hidx[t] = bi;
}

// ---------------------------------------------------------------------------
// Kernel 4: top-3 over l of R2[l,m] = R[H_idx[l], m]  (gather instead of 2nd GEMM)
// Writes S directly to d_out and final_idx = H_idx[hard].
// ---------------------------------------------------------------------------
__global__ void topk_kernel(const float* __restrict__ R, const int* __restrict__ Hidx,
                            float* __restrict__ Sout, int* __restrict__ fidx)
{
    const int t = blockIdx.x * 256 + threadIdx.x;
    if (t >= BSZ * LDIM) return;
    const int b = t / LDIM, m = t % LDIM;
    const float* Rb = R + (size_t)b * LDIM * LDIM;
    const int* Hb = Hidx + b * LDIM;
    float v0 = -INFINITY, v1 = -INFINITY, v2 = -INFINITY;
    int h0 = 0, h1 = 0, h2 = 0;
    for (int l = 0; l < LDIM; ++l) {
        const int hl = Hb[l];                    // scalar-broadcast across the wave
        const float v = Rb[(size_t)hl * LDIM + m];
        if (v > v0)      { v2 = v1; h2 = h1; v1 = v0; h1 = h0; v0 = v; h0 = hl; }
        else if (v > v1) { v2 = v1; h2 = h1; v1 = v;  h1 = hl; }
        else if (v > v2) { v2 = v;  h2 = hl; }
    }
    Sout[(0 * BSZ + b) * LDIM + m] = v0;
    Sout[(1 * BSZ + b) * LDIM + m] = v1;
    Sout[(2 * BSZ + b) * LDIM + m] = v2;
    fidx[(0 * BSZ + b) * LDIM + m] = h0;
    fidx[(1 * BSZ + b) * LDIM + m] = h1;
    fidx[(2 * BSZ + b) * LDIM + m] = h2;
}

// ---------------------------------------------------------------------------
// Kernels 5-7: gather + fold.  fold(gather(unfold(ref), fidx)) / 9, expressed
// directly: each output pixel sums its 9 live (di,dj) taps.
// ---------------------------------------------------------------------------
__global__ void fold3_kernel(const float* __restrict__ ref3, const int* __restrict__ fidx,
                             float* __restrict__ out)
{
    const int t = blockIdx.x * 256 + threadIdx.x;         // (3,2,256,48,48)
    const int x = t % 48, y = (t / 48) % 48, c = (t / 2304) % 256;
    const int bi = t / (2304 * 256), b = bi % 2, i = bi / 2;
    const int* fx = fidx + (i * BSZ + b) * LDIM;
    const float* rc = ref3 + (size_t)(b * 256 + c) * 2304;
    float acc = 0.f;
#pragma unroll
    for (int di = 0; di < 3; ++di) {
        const int oy = y + 1 - di;
        if ((unsigned)oy >= 48u) continue;
#pragma unroll
        for (int dj = 0; dj < 3; ++dj) {
            const int ox = x + 1 - dj;
            if ((unsigned)ox >= 48u) continue;
            const int mp = fx[oy * 48 + ox];
            const int py = mp / 48 + di - 1;
            const int px = mp % 48 + dj - 1;
            if ((unsigned)py < 48u && (unsigned)px < 48u) acc += rc[py * 48 + px];
        }
    }
    out[t] = acc * (1.f / 9.f);
}

__global__ void fold2_kernel(const float* __restrict__ ref2, const int* __restrict__ fidx,
                             float* __restrict__ out)
{
    const int t = blockIdx.x * 256 + threadIdx.x;         // (3,2,128,96,96)
    const int x = t % 96, y = (t / 96) % 96, c = (t / (96 * 96)) % 128;
    const int bi = t / (96 * 96 * 128), b = bi % 2, i = bi / 2;
    const int* fx = fidx + (i * BSZ + b) * LDIM;
    const float* rc = ref2 + (size_t)(b * 128 + c) * 96 * 96;
    float acc = 0.f;
    const int d0y = y & 1, d0x = x & 1;                   // only di == y (mod 2) live
#pragma unroll
    for (int a = 0; a < 3; ++a) {
        const int di = d0y + a * 2;
        const int ty = y + 2 - di;
        if (ty < 0) continue;
        const int oy = ty >> 1;
        if (oy >= 48) continue;
#pragma unroll
        for (int e = 0; e < 3; ++e) {
            const int dj = d0x + e * 2;
            const int tx = x + 2 - dj;
            if (tx < 0) continue;
            const int ox = tx >> 1;
            if (ox >= 48) continue;
            const int mp = fx[oy * 48 + ox];
            const int sy = (mp / 48) * 2 + di - 2;
            const int sx = (mp % 48) * 2 + dj - 2;
            if ((unsigned)sy < 96u && (unsigned)sx < 96u) acc += rc[sy * 96 + sx];
        }
    }
    out[t] = acc * (1.f / 9.f);
}

__global__ void fold1_kernel(const float* __restrict__ ref1, const int* __restrict__ fidx,
                             float* __restrict__ out)
{
    const int t = blockIdx.x * 256 + threadIdx.x;         // (3,2,64,192,192)
    const int x = t % 192, y = (t / 192) % 192, c = (t / (192 * 192)) % 64;
    const int bi = t / (192 * 192 * 64), b = bi % 2, i = bi / 2;
    const int* fx = fidx + (i * BSZ + b) * LDIM;
    const float* rc = ref1 + (size_t)(b * 64 + c) * 192 * 192;
    float acc = 0.f;
    const int d0y = y & 3, d0x = x & 3;                   // only di == y (mod 4) live
#pragma unroll
    for (int a = 0; a < 3; ++a) {
        const int di = d0y + a * 4;
        const int ty = y + 4 - di;
        if (ty < 0) continue;
        const int oy = ty >> 2;
        if (oy >= 48) continue;
#pragma unroll
        for (int e = 0; e < 3; ++e) {
            const int dj = d0x + e * 4;
            const int tx = x + 4 - dj;
            if (tx < 0) continue;
            const int ox = tx >> 2;
            if (ox >= 48) continue;
            const int mp = fx[oy * 48 + ox];
            const int sy = (mp / 48) * 4 + di - 4;
            const int sx = (mp % 48) * 4 + dj - 4;
            if ((unsigned)sy < 192u && (unsigned)sx < 192u) acc += rc[sy * 192 + sx];
        }
    }
    out[t] = acc * (1.f / 9.f);
}

// ---------------------------------------------------------------------------
extern "C" void kernel_launch(void* const* d_in, const int* in_sizes, int n_in,
                              void* d_out, int out_size, void* d_ws, size_t ws_size,
                              hipStream_t stream)
{
    const float* lr_lv3    = (const float*)d_in[0];
    const float* refsr_lv3 = (const float*)d_in[1];
    const float* ref_lv1   = (const float*)d_in[2];
    const float* ref_lv2   = (const float*)d_in[3];
    const float* ref_lv3   = (const float*)d_in[4];
    float* out = (float*)d_out;

    // workspace layout (85,008,384 bytes total)
    char* ws = (char*)d_ws;
    _Float16* QnT = (_Float16*)ws;                               // 21,233,664 B
    _Float16* KnT = (_Float16*)(ws + 21233664ull);               // 21,233,664 B
    float*    R   = (float*)  (ws + 42467328ull);                // 42,467,328 B
    int*      Hidx= (int*)    (ws + 84934656ull);                //     18,432 B
    int*      fidx= (int*)    (ws + 84953088ull);                //     55,296 B

    // output region offsets (floats)
    float* S_out  = out;                        // 3*2*2304            = 13,824
    float* T3_out = out + 13824;                // 3*2*256*48*48       = 3,538,944
    float* T2_out = out + 3552768;              // 3*2*128*96*96       = 7,077,888
    float* T1_out = out + 10630656;             // 3*2*64*192*192      = 14,155,776

    unfold_norm_kernel<<<BSZ * LDIM, 256, 0, stream>>>(lr_lv3, refsr_lv3, QnT, KnT);

    dim3 gg(LDIM / 64, LDIM / 256, BSZ);        // 36 x 9 x 2
    corr_gemm_kernel<<<gg, 256, 0, stream>>>(KnT, QnT, R);

    argmax_kernel<<<(BSZ * LDIM) / 256, 256, 0, stream>>>(R, Hidx);
    topk_kernel  <<<(BSZ * LDIM) / 256, 256, 0, stream>>>(R, Hidx, S_out, fidx);

    fold3_kernel<<< 3538944 / 256, 256, 0, stream>>>(ref_lv3, fidx, T3_out);
    fold2_kernel<<< 7077888 / 256, 256, 0, stream>>>(ref_lv2, fidx, T2_out);
    fold1_kernel<<<14155776 / 256, 256, 0, stream>>>(ref_lv1, fidx, T1_out);
}
